// SlidingWindowAttention_53455162966289
// MI455X (gfx1250) — compile-verified, matrix operand order checked by reading, not run on hardware
//
#include <hip/hip_runtime.h>

// ---------------------------------------------------------------------------
// Types
// ---------------------------------------------------------------------------
typedef __bf16 bf16;
typedef __attribute__((ext_vector_type(8)))  __bf16   v8bf;
typedef __attribute__((ext_vector_type(16))) __bf16   v16bf;
typedef __attribute__((ext_vector_type(8)))  float    v8f;
typedef __attribute__((ext_vector_type(4)))  unsigned v4u;
typedef __attribute__((ext_vector_type(4)))  int      v4i;
typedef __attribute__((ext_vector_type(8)))  int      v8i;

#if defined(__has_builtin)
#if __has_builtin(__builtin_amdgcn_tensor_load_to_lds) && \
    __has_builtin(__builtin_amdgcn_s_wait_tensorcnt)
#define CDNA5_HAS_TDM 1
#endif
#endif
#ifndef CDNA5_HAS_TDM
#define CDNA5_HAS_TDM 0
#endif

// f32 -> bf16 round-to-nearest-even (storage-only bf16; no bf16 VALU needed)
__device__ __forceinline__ bf16 f2bf(float f) {
  unsigned u = __float_as_uint(f);
  unsigned r = u + 0x7FFFu + ((u >> 16) & 1u);
  unsigned short h = (unsigned short)(r >> 16);
  bf16 out;
  __builtin_memcpy(&out, &h, 2);
  return out;
}

// Load one 16-bit A/B WMMA fragment (CDNA5 layout):
//   elements 0..7  = p[0..7]    (this lane's low-K block, koff pre-applied)
//   elements 8..15 = p[16..23]  (this lane's high-K block)
__device__ __forceinline__ v16bf load_frag(const bf16* p) {
  v8bf lo = *(const v8bf*)p;
  v8bf hi = *(const v8bf*)(p + 16);
  return __builtin_shufflevector(lo, hi, 0, 1, 2, 3, 4, 5, 6, 7,
                                         8, 9, 10, 11, 12, 13, 14, 15);
}

__device__ __forceinline__ v8f wmma_bf16(v16bf a, v16bf b, v8f c) {
  // (neg_a, A, neg_b, B, c_mod, C, reuse_a, reuse_b)
  return __builtin_amdgcn_wmma_f32_16x16x32_bf16(false, a, false, b, (short)0, c,
                                                 false, false);
}

// 16-lane (half-wave) reductions; xor masks 1,2,4,8 never cross the lane-16 boundary
__device__ __forceinline__ float rmax16(float v) {
  v = fmaxf(v, __shfl_xor(v, 1, 32));
  v = fmaxf(v, __shfl_xor(v, 2, 32));
  v = fmaxf(v, __shfl_xor(v, 4, 32));
  v = fmaxf(v, __shfl_xor(v, 8, 32));
  return v;
}
__device__ __forceinline__ float rsum16(float v) {
  v += __shfl_xor(v, 1, 32);
  v += __shfl_xor(v, 2, 32);
  v += __shfl_xor(v, 4, 32);
  v += __shfl_xor(v, 8, 32);
  return v;
}

// ---------------------------------------------------------------------------
// TDM: issue a 2-D tile load (tile0 x tile1 elements of 2 bytes) from a
// row-major tensor (row stride = dim0 elements) into contiguous LDS.
// Descriptor packing per CDNA5 ISA.
// Caller must be a single wave; follow with s_wait_tensorcnt + barrier.
// ---------------------------------------------------------------------------
__device__ __forceinline__ void tdm_load_tile_2d(const bf16* gtile, unsigned lds_off,
                                                 unsigned dim0, unsigned dim1,
                                                 unsigned tile0, unsigned tile1) {
#if CDNA5_HAS_TDM
  unsigned long long ga = (unsigned long long)(uintptr_t)gtile;
  // Group 0: count=1 | lds_addr | global_addr[56:0] | type=2
  v4u g0 = {1u, lds_off, (unsigned)(ga & 0xFFFFFFFFu),
            (unsigned)((ga >> 32) & 0x01FFFFFFu) | (2u << 30)};
  // Group 1: data_size=1 (2B) at [17:16]; tensor_dim0 at [79:48];
  // tensor_dim1 at [111:80]; tile_dim0 at [127:112]; tile_dim1 at [143:128];
  // tensor_dim0_stride (48b) at [207:160].
  v8i g1 = {(int)0x00010000u,
            (int)((dim0 & 0xFFFFu) << 16),
            (int)((dim0 >> 16) | ((dim1 & 0xFFFFu) << 16)),
            (int)((dim1 >> 16) | (tile0 << 16)),
            (int)tile1,
            (int)dim0,  // stride = row pitch = dim0 elements
            0, 0};
  v4i gz = {0, 0, 0, 0};
#if __clang_major__ >= 23
  v8i gz8 = {0, 0, 0, 0, 0, 0, 0, 0};
  __builtin_amdgcn_tensor_load_to_lds(g0, g1, gz, gz, gz8, 0);
#else
  __builtin_amdgcn_tensor_load_to_lds(g0, g1, gz, gz, 0);
#endif
#else
  (void)gtile; (void)lds_off; (void)dim0; (void)dim1; (void)tile0; (void)tile1;
#endif
}

// ---------------------------------------------------------------------------
// Kernel 1: f32 -> bf16 elementwise convert
// ---------------------------------------------------------------------------
__global__ void f32_to_bf16_k(const float* __restrict__ src, bf16* __restrict__ dst,
                              int n) {
  int i = blockIdx.x * blockDim.x + threadIdx.x;
  if (i < n) dst[i] = f2bf(src[i]);
}

// ---------------------------------------------------------------------------
// Kernel 2: bf16 GEMM  C[M,N] = A[M,K] * B[K,N], f32 accumulate.
// Block: 256 threads (8 waves), 128x128 tile, BK=32.
// A tile staged via the Tensor Data Mover (wave 0 issues, TENSORcnt wait);
// B tile staged transposed in LDS by all threads. Wave grid 4x2; each wave
// computes 32x64 via 2x4 WMMA 16x16x32 accumulators.
// ---------------------------------------------------------------------------
template <bool OUT_BF16>
__global__ __launch_bounds__(256) void gemm_bf16_k(const bf16* __restrict__ A,
                                                   const bf16* __restrict__ B,
                                                   void* __restrict__ Cp,
                                                   int M, int N, int Kd) {
  __shared__ bf16 As[128 * 32];  // [row][k], row-major
  __shared__ bf16 Bs[128 * 32];  // [col][k], i.e. B transposed

  const int t   = threadIdx.x;
  const int lid = t & 31;
  const int wv  = t >> 5;
  const int lr  = lid & 15;
  const int lh  = lid >> 4;
  const int wm  = (wv >> 1) * 32;  // wave M offset within block tile
  const int wn  = (wv & 1) * 64;   // wave N offset within block tile
  const int m0  = blockIdx.y * 128;
  const int n0  = blockIdx.x * 128;

  v8f acc[2][4] = {};

  const int ar = t >> 1, ac = (t & 1) * 16;   // A loader map (fallback path)
  const int bk = t >> 3, bc = (t & 7) * 16;   // B loader map: 32 rows x 128 cols

  const unsigned as_off = (unsigned)(uintptr_t)(void*)As;  // LDS byte offset

  for (int k0 = 0; k0 < Kd; k0 += 32) {
    // B tile global load (16 bf16 per thread = 2 x b128)
    const bf16* bp = B + (size_t)(k0 + bk) * N + n0 + bc;
    v8bf b0 = *(const v8bf*)bp;
    v8bf b1 = *(const v8bf*)(bp + 8);
    if (k0 + 32 < Kd)  // prefetch next B tile (-> global_prefetch_b8)
      __builtin_prefetch(bp + (size_t)32 * N, 0, 1);

#if !CDNA5_HAS_TDM
    const bf16* ap = A + (size_t)(m0 + ar) * Kd + k0 + ac;
    v8bf a0 = *(const v8bf*)ap;
    v8bf a1 = *(const v8bf*)(ap + 8);
#endif

    __syncthreads();  // previous iteration's fragment reads are done

#if CDNA5_HAS_TDM
    if (wv == 0)  // one wave drives the Tensor Data Mover for the A tile
      tdm_load_tile_2d(A + (size_t)m0 * Kd + k0, as_off,
                       (unsigned)Kd, (unsigned)M, 32u, 128u);
#else
    *(v8bf*)&As[ar * 32 + ac]     = a0;
    *(v8bf*)&As[ar * 32 + ac + 8] = a1;
#endif
#pragma unroll
    for (int q = 0; q < 8; ++q) {
      Bs[(bc + q) * 32 + bk]     = b0[q];
      Bs[(bc + 8 + q) * 32 + bk] = b1[q];
    }
#if CDNA5_HAS_TDM
    if (wv == 0) __builtin_amdgcn_s_wait_tensorcnt(0);
#endif
    __syncthreads();

    v16bf af[2], bfr[4];
#pragma unroll
    for (int tm = 0; tm < 2; ++tm)
      af[tm] = load_frag(&As[(wm + tm * 16 + lr) * 32 + lh * 8]);
#pragma unroll
    for (int tn = 0; tn < 4; ++tn)
      bfr[tn] = load_frag(&Bs[(wn + tn * 16 + lr) * 32 + lh * 8]);
#pragma unroll
    for (int tm = 0; tm < 2; ++tm)
#pragma unroll
      for (int tn = 0; tn < 4; ++tn)
        acc[tm][tn] = wmma_bf16(af[tm], bfr[tn], acc[tm][tn]);
  }

  // C layout: VGPR r, lanes 0-15 -> M=r, lanes 16-31 -> M=8+r; N = lane%16
#pragma unroll
  for (int tm = 0; tm < 2; ++tm)
#pragma unroll
    for (int tn = 0; tn < 4; ++tn)
#pragma unroll
      for (int r = 0; r < 8; ++r) {
        int row = m0 + wm + tm * 16 + lh * 8 + r;
        int col = n0 + wn + tn * 16 + lr;
        float v = acc[tm][tn][r];
        if (OUT_BF16)
          ((bf16*)Cp)[(size_t)row * N + col] = f2bf(v);
        else
          ((float*)Cp)[(size_t)row * N + col] = v;
      }
}

// ---------------------------------------------------------------------------
// Kernel 3: sliding-window attention, flash style.
// One wave (32 threads) per (16-query tile, head). Dh = 64, window = 256.
// Key chunks of 32; 9 chunks cover [q0-256, q0+31].
// ---------------------------------------------------------------------------
__global__ __launch_bounds__(32) void swa16_k(const bf16* __restrict__ Q,
                                              const bf16* __restrict__ Kx,
                                              const bf16* __restrict__ V,
                                              bf16* __restrict__ O) {
  __shared__ bf16 Pl[16 * 32];  // probs, [row][key-in-chunk]
  __shared__ bf16 Vt[64 * 32];  // V chunk transposed, [dim][key-in-chunk]

  const int lid = threadIdx.x;
  const int lr  = lid & 15;
  const int lh  = lid >> 4;
  const int q0  = blockIdx.x * 16;
  const int h   = blockIdx.y;

  const bf16* Qh = Q + h * 64;
  const bf16* Kh = Kx + h * 64;
  const bf16* Vh = V + h * 64;

  v8f o[4] = {};
  float m[8], l[8];
#pragma unroll
  for (int r = 0; r < 8; ++r) { m[r] = -3.0e38f; l[r] = 0.0f; }

  for (int c = 0; c < 9; ++c) {
    const int jbase = q0 - 256 + 32 * c;
    if (jbase + 31 < 0) continue;  // chunk fully out of range (uniform)

    const int j0  = jbase + lr;
    const int j1  = jbase + 16 + lr;
    const int jc0 = min(max(j0, 0), 4095);
    const int jc1 = min(max(j1, 0), 4095);

    // prefetch next chunk's V rows (one row per lane)
    if (c < 8) {
      int jn = min(max(jbase + 32 + lid, 0), 4095);
      __builtin_prefetch(Vh + (size_t)jn * 1024, 0, 1);
    }

    // S = Q * K^T over Dh=64 (two 32-wide K-steps). Row-major K rows are the
    // column-major B operand, so fragments load contiguously from global.
    v8f s0 = {}, s1 = {};
#pragma unroll
    for (int ks = 0; ks < 2; ++ks) {
      const int d0 = ks * 32 + lh * 8;
      v16bf qf  = load_frag(Qh + (size_t)(q0 + lr) * 1024 + d0);
      v16bf k0f = load_frag(Kh + (size_t)jc0 * 1024 + d0);
      v16bf k1f = load_frag(Kh + (size_t)jc1 * 1024 + d0);
      s0 = wmma_bf16(qf, k0f, s0);
      s1 = wmma_bf16(qf, k1f, s1);
    }

    // mask + online softmax (per-row stats, 16-lane reductions)
#pragma unroll
    for (int r = 0; r < 8; ++r) {
      const int i = q0 + lh * 8 + r;
      float a0 = s0[r] * 0.125f;  // 1/sqrt(64)
      float a1 = s1[r] * 0.125f;
      bool ok0 = (j0 >= 0) && (j0 <= i) && (j0 >= i - 256);
      bool ok1 = (j1 >= 0) && (j1 <= i) && (j1 >= i - 256);
      a0 = ok0 ? a0 : -3.0e38f;
      a1 = ok1 ? a1 : -3.0e38f;
      float mt   = rmax16(fmaxf(a0, a1));
      float mn   = fmaxf(m[r], mt);
      float corr = __expf(m[r] - mn);
      float p0   = __expf(a0 - mn);
      float p1   = __expf(a1 - mn);
      float rs   = rsum16(p0 + p1);
      l[r] = l[r] * corr + rs;
      m[r] = mn;
#pragma unroll
      for (int tn = 0; tn < 4; ++tn) o[tn][r] *= corr;
      s0[r] = p0;
      s1[r] = p1;
    }

    // stage P (accumulator layout -> row-major 16x32 in LDS)
#pragma unroll
    for (int r = 0; r < 8; ++r) {
      const int row = lh * 8 + r;
      Pl[row * 32 + lr]      = f2bf(s0[r]);
      Pl[row * 32 + 16 + lr] = f2bf(s1[r]);
    }
    // stage V chunk transposed: Vt[dim][key]; lane = one key row
    {
      const int j  = jbase + lid;
      const int jc = min(max(j, 0), 4095);
      const bf16* vp = Vh + (size_t)jc * 1024;
#pragma unroll
      for (int d = 0; d < 64; ++d) Vt[d * 32 + lid] = vp[d];
    }
    __syncthreads();

    v16bf pf = load_frag(&Pl[lr * 32 + lh * 8]);
    v16bf vf[4];
#pragma unroll
    for (int tn = 0; tn < 4; ++tn)
      vf[tn] = load_frag(&Vt[(tn * 16 + lr) * 32 + lh * 8]);
    __syncthreads();  // fragments in registers; LDS reusable next chunk

#pragma unroll
    for (int tn = 0; tn < 4; ++tn) o[tn] = wmma_bf16(pf, vf[tn], o[tn]);
  }

  // normalize and write O (pre-Wo activations), bf16 [S, 1024]
#pragma unroll
  for (int tn = 0; tn < 4; ++tn)
#pragma unroll
    for (int r = 0; r < 8; ++r) {
      float v = o[tn][r] / l[r];
      int row = q0 + lh * 8 + r;
      int col = h * 64 + tn * 16 + lr;
      O[(size_t)row * 1024 + col] = f2bf(v);
    }
}

// ---------------------------------------------------------------------------
// Launcher
// ---------------------------------------------------------------------------
extern "C" void kernel_launch(void* const* d_in, const int* in_sizes, int n_in,
                              void* d_out, int out_size, void* d_ws, size_t ws_size,
                              hipStream_t stream) {
  (void)in_sizes; (void)n_in; (void)out_size; (void)ws_size;

  const float* x  = (const float*)d_in[0];  // [4096, 1024]
  const float* Wq = (const float*)d_in[1];  // [1024, 1024]
  const float* Wk = (const float*)d_in[2];
  const float* Wv = (const float*)d_in[3];
  const float* Wo = (const float*)d_in[4];

  char* w = (char*)d_ws;  // workspace layout (48 MB total)
  bf16* xb  = (bf16*)(w + (size_t)(0 << 20));
  bf16* wqb = (bf16*)(w + (size_t)(8 << 20));
  bf16* wkb = (bf16*)(w + (size_t)(10 << 20));
  bf16* wvb = (bf16*)(w + (size_t)(12 << 20));
  bf16* wob = (bf16*)(w + (size_t)(14 << 20));
  bf16* Qb  = (bf16*)(w + (size_t)(16 << 20));
  bf16* Kb  = (bf16*)(w + (size_t)(24 << 20));
  bf16* Vb  = (bf16*)(w + (size_t)(32 << 20));
  bf16* Ob  = (bf16*)(w + (size_t)(40 << 20));

  const int nX = 4096 * 1024;
  const int nW = 1024 * 1024;
  f32_to_bf16_k<<<nX / 256, 256, 0, stream>>>(x, xb, nX);
  f32_to_bf16_k<<<nW / 256, 256, 0, stream>>>(Wq, wqb, nW);
  f32_to_bf16_k<<<nW / 256, 256, 0, stream>>>(Wk, wkb, nW);
  f32_to_bf16_k<<<nW / 256, 256, 0, stream>>>(Wv, wvb, nW);
  f32_to_bf16_k<<<nW / 256, 256, 0, stream>>>(Wo, wob, nW);

  dim3 gg(1024 / 128, 4096 / 128);  // (N tiles, M tiles)
  gemm_bf16_k<true><<<gg, 256, 0, stream>>>(xb, wqb, Qb, 4096, 1024, 1024);
  gemm_bf16_k<true><<<gg, 256, 0, stream>>>(xb, wkb, Kb, 4096, 1024, 1024);
  gemm_bf16_k<true><<<gg, 256, 0, stream>>>(xb, wvb, Vb, 4096, 1024, 1024);

  swa16_k<<<dim3(4096 / 16, 16), 32, 0, stream>>>(Qb, Kb, Vb, Ob);

  gemm_bf16_k<false><<<gg, 256, 0, stream>>>(Ob, wob, d_out, 4096, 1024, 1024);
}
